// ConsistencyConstraint_15504831939184
// MI455X (gfx1250) — matrix operand order repaired and checked by reading, not compiled, not run on hardware
//
#include <hip/hip_runtime.h>
#include <hip/hip_bf16.h>

typedef __attribute__((ext_vector_type(8)))  float   v8f;
typedef __attribute__((ext_vector_type(16))) __bf16  v16bf;
typedef __attribute__((ext_vector_type(8)))  __bf16  v8bf;
typedef __attribute__((ext_vector_type(4)))  int     v4i;

// Address-space-qualified pointer types for the async DMA builtin
// (param 1: int4 in AS1/global, param 2: int4 in AS3/LDS).
typedef __attribute__((address_space(1))) v4i* as1_v4i_p;
typedef __attribute__((address_space(3))) v4i* as3_v4i_p;

#define NB    2048
#define KD    4096
#define NCLS  10
#define TILE  128
#define NT    (NB / TILE)              // 16 tile rows/cols
#define NPAIR (NT * (NT + 1) / 2)      // 136 upper-triangular tile pairs

#define KSTEP 64                       // K slice staged in LDS per iteration
#define ROWP  (KSTEP + 8)              // padded row stride (halves) -> 144B, kills bank conflicts
#define TSZ   (TILE * ROWP)            // halves per staged tile
#define LDS_BYTES (2 * 4 * TSZ * (int)sizeof(__bf16))   // 2 buffers x 4 tiles = 147456 B

// ---------------------------------------------------------------------------
// CDNA5 async global->LDS copy (ASYNCcnt) with safe fallback.
// ---------------------------------------------------------------------------
__device__ __forceinline__ void copy16_g2l(const __bf16* g, __bf16* l) {
#if __has_builtin(__builtin_amdgcn_global_load_async_to_lds_b128)
    __builtin_amdgcn_global_load_async_to_lds_b128(
        (as1_v4i_p)(__bf16*)g, (as3_v4i_p)l, 0, 0);
#else
    *(v8bf*)l = *(const v8bf*)g;       // load-to-VGPR + ds_store fallback
#endif
}

__device__ __forceinline__ void async_wait0() {
#if __has_builtin(__builtin_amdgcn_global_load_async_to_lds_b128) && \
    __has_builtin(__builtin_amdgcn_s_wait_asynccnt)
    __builtin_amdgcn_s_wait_asynccnt(0);
#endif
}

// ---------------------------------------------------------------------------
// Zero the scalar output (harness poisons d_out; we accumulate atomically).
// ---------------------------------------------------------------------------
__global__ void zero_out_kernel(float* out) {
    if (threadIdx.x == 0) out[0] = 0.0f;
}

// ---------------------------------------------------------------------------
// Prep: per-row min-max normalize grad -> G (bf16), copy x -> X (bf16),
// exact f32 row norms, argmax(outputs) -> pred, CE contribution -> out.
// One 256-thread block per sample row (4096 elements, 16 per thread).
// ---------------------------------------------------------------------------
__global__ __launch_bounds__(256) void prep_kernel(
    const float* __restrict__ outputs, const float* __restrict__ grad,
    const float* __restrict__ x,       const int*   __restrict__ y,
    __bf16* __restrict__ Gbf, __bf16* __restrict__ Xbf,
    float*  __restrict__ normG, float* __restrict__ normX,
    int*    __restrict__ pred, float*  __restrict__ out)
{
    const int b = blockIdx.x;
    const int t = threadIdx.x;
    const float* grow = grad + (size_t)b * KD;
    const float* xrow = x    + (size_t)b * KD;

    float r[16];
    float mn = 1e30f, mx = -1e30f;
    #pragma unroll
    for (int u = 0; u < 16; ++u) {
        float v = grow[t + 256 * u];
        r[u] = v;
        mn = fminf(mn, v);
        mx = fmaxf(mx, v);
    }

    __shared__ float red[256];
    red[t] = mn; __syncthreads();
    for (int s = 128; s > 0; s >>= 1) { if (t < s) red[t] = fminf(red[t], red[t + s]); __syncthreads(); }
    mn = red[0]; __syncthreads();
    red[t] = mx; __syncthreads();
    for (int s = 128; s > 0; s >>= 1) { if (t < s) red[t] = fmaxf(red[t], red[t + s]); __syncthreads(); }
    mx = red[0]; __syncthreads();

    const float inv = 1.0f / (mx - mn);
    float ssG = 0.0f, ssX = 0.0f;
    #pragma unroll
    for (int u = 0; u < 16; ++u) {
        const int k = t + 256 * u;
        float gv = (r[u] - mn) * inv;
        ssG += gv * gv;
        Gbf[(size_t)b * KD + k] = (__bf16)gv;
        float xv = xrow[k];
        ssX += xv * xv;
        Xbf[(size_t)b * KD + k] = (__bf16)xv;
    }

    red[t] = ssG; __syncthreads();
    for (int s = 128; s > 0; s >>= 1) { if (t < s) red[t] += red[t + s]; __syncthreads(); }
    if (t == 0) normG[b] = sqrtf(red[0]);
    __syncthreads();
    red[t] = ssX; __syncthreads();
    for (int s = 128; s > 0; s >>= 1) { if (t < s) red[t] += red[t + s]; __syncthreads(); }
    if (t == 0) normX[b] = sqrtf(red[0]);

    if (t == 0) {
        const float* o = outputs + (size_t)b * NCLS;
        float m = o[0]; int am = 0;
        #pragma unroll
        for (int c = 1; c < NCLS; ++c) { float v = o[c]; if (v > m) { m = v; am = c; } }
        pred[b] = am;
        float s = 0.0f;
        #pragma unroll
        for (int c = 0; c < NCLS; ++c) s += __expf(o[c] - m);
        float lse = m + __logf(s);
        float ce  = lse - o[y[b]];
        atomicAdd(out, ce * (1.0f / NB));
    }
}

// ---------------------------------------------------------------------------
// Stage a 128 x KSTEP bf16 slice of all four operand tiles into LDS.
// 256 threads x 16B x 4 passes per tile; async DMA when available.
// ---------------------------------------------------------------------------
__device__ __forceinline__ void stage_tiles(
    const __bf16* __restrict__ G, const __bf16* __restrict__ X,
    int iBase, int jBase, int k0, __bf16* sm, int t)
{
    const __bf16* src0 = G + (size_t)iBase * KD;
    const __bf16* src1 = G + (size_t)jBase * KD;
    const __bf16* src2 = X + (size_t)iBase * KD;
    const __bf16* src3 = X + (size_t)jBase * KD;
    const __bf16* src[4] = { src0, src1, src2, src3 };
    #pragma unroll
    for (int tile = 0; tile < 4; ++tile) {
        __bf16* dst = sm + tile * TSZ;
        #pragma unroll
        for (int p = 0; p < 4; ++p) {
            const int q   = p * 256 + t;      // 0..1023
            const int row = q >> 3;           // 0..127
            const int kc  = (q & 7) * 8;      // 0..56 (halves)
            copy16_g2l(src[tile] + (size_t)row * KD + k0 + kc,
                       dst + row * ROWP + kc);
        }
    }
}

// ---------------------------------------------------------------------------
// Fused triangular Gram + masked ratio reduction.
// 136 blocks over upper-triangular 128x128 tile pairs; double-buffered LDS
// staging (async global->LDS), 8 waves each computing a 32x64 patch of both
// Cg and Cx with bf16 WMMA (f32 accumulate), then the masked reduction.
// ---------------------------------------------------------------------------
__global__ __launch_bounds__(256) void pair_kernel(
    const __bf16* __restrict__ G, const __bf16* __restrict__ X,
    const float* __restrict__ normG, const float* __restrict__ normX,
    const int*   __restrict__ pred,  float* __restrict__ out)
{
    extern __shared__ __bf16 smem[];           // 2 buffers x 4 tiles x TSZ

    // Decode upper-triangular tile pair (ti <= tj).
    int idx = blockIdx.x, ti = 0;
    while (idx >= NT - ti) { idx -= NT - ti; ++ti; }
    const int tj = ti + idx;
    const int iBase = ti * TILE;
    const int jBase = tj * TILE;

    const int t    = threadIdx.x;
    const int wave = t >> 5;
    const int lane = t & 31;
    const int wm = wave & 3;      // 4 row-groups of 32
    const int wn = wave >> 2;     // 2 col-groups of 64
    const int hi = lane >> 4;     // K-half select: element s holds k = 16*hi + s in A and B (contiguous order)
    const int lr = lane & 15;

    union AF { v16bf v; v8bf h[2]; };

    v8f cg[2][4] = {};
    v8f cx[2][4] = {};

    __bf16* buf0 = smem;
    __bf16* buf1 = smem + 4 * TSZ;

    stage_tiles(G, X, iBase, jBase, 0, buf0, t);   // prologue fill

    for (int k0 = 0; k0 < KD; k0 += KSTEP) {
        __bf16* cur = ((k0 / KSTEP) & 1) ? buf1 : buf0;
        __bf16* nxt = ((k0 / KSTEP) & 1) ? buf0 : buf1;

        async_wait0();            // current buffer's DMA complete (no-op in fallback)
        __syncthreads();          // ... and visible to all waves

        if (k0 + KSTEP < KD)      // prefetch next slice; overlaps the WMMAs below
            stage_tiles(G, X, iBase, jBase, k0 + KSTEP, nxt, t);

        const __bf16* sGi = cur + 0 * TSZ;
        const __bf16* sGj = cur + 1 * TSZ;
        const __bf16* sXi = cur + 2 * TSZ;
        const __bf16* sXj = cur + 3 * TSZ;

        #pragma unroll
        for (int kk = 0; kk < KSTEP; kk += 32) {
            AF aG[2], aX[2], bG[4], bX[4];
            #pragma unroll
            for (int tm = 0; tm < 2; ++tm) {
                const int ar = (wm * 32 + tm * 16 + lr) * ROWP + kk;
                aG[tm].h[0] = *(const v8bf*)(sGi + ar + hi * 16);
                aG[tm].h[1] = *(const v8bf*)(sGi + ar + hi * 16 + 8);
                aX[tm].h[0] = *(const v8bf*)(sXi + ar + hi * 16);
                aX[tm].h[1] = *(const v8bf*)(sXi + ar + hi * 16 + 8);
            }
            #pragma unroll
            for (int tn = 0; tn < 4; ++tn) {
                const int br = (wn * 64 + tn * 16 + lr) * ROWP + kk + hi * 16;
                bG[tn].h[0] = *(const v8bf*)(sGj + br);
                bG[tn].h[1] = *(const v8bf*)(sGj + br + 8);
                bX[tn].h[0] = *(const v8bf*)(sXj + br);
                bX[tn].h[1] = *(const v8bf*)(sXj + br + 8);
            }
            #pragma unroll
            for (int tm = 0; tm < 2; ++tm) {
                #pragma unroll
                for (int tn = 0; tn < 4; ++tn) {
                    cg[tm][tn] = __builtin_amdgcn_wmma_f32_16x16x32_bf16(
                        false, aG[tm].v, false, bG[tn].v, (short)0, cg[tm][tn], false, false);
                    cx[tm][tn] = __builtin_amdgcn_wmma_f32_16x16x32_bf16(
                        false, aX[tm].v, false, bX[tn].v, (short)0, cx[tm][tn], false, false);
                }
            }
        }
        __syncthreads();          // all reads of `cur` done before it is restaged
    }

    // Epilogue: C/D layout -> M = r + 8*(lane>>4), N = lane&15.
    const int m0 = iBase + wm * 32;
    const int n0 = jBase + wn * 64;
    float acc = 0.0f;
    #pragma unroll
    for (int tm = 0; tm < 2; ++tm) {
        #pragma unroll
        for (int tn = 0; tn < 4; ++tn) {
            #pragma unroll
            for (int r = 0; r < 8; ++r) {
                const int i = m0 + tm * 16 + r + 8 * hi;
                const int j = n0 + tn * 16 + lr;
                if (i < j && pred[i] == pred[j]) {
                    float cgv = cg[tm][tn][r] / fmaxf(normG[i] * normG[j], 1e-8f);
                    float cxv = cx[tm][tn][r] / fmaxf(normX[i] * normX[j], 1e-8f);
                    acc += (1.0f - cgv) / (1.0f - cxv);
                }
            }
        }
    }
    #pragma unroll
    for (int off = 16; off > 0; off >>= 1) acc += __shfl_xor(acc, off, 32);
    if (lane == 0) atomicAdd(out, acc * (1.0f / NB));
}

// ---------------------------------------------------------------------------
extern "C" void kernel_launch(void* const* d_in, const int* in_sizes, int n_in,
                              void* d_out, int out_size, void* d_ws, size_t ws_size,
                              hipStream_t stream) {
    (void)in_sizes; (void)n_in; (void)out_size; (void)ws_size;
    const float* outputs = (const float*)d_in[0];   // [2048, 10]
    const float* grad    = (const float*)d_in[1];   // [2048, 1, 64, 64]
    const float* x       = (const float*)d_in[2];   // [2048, 1, 64, 64]
    const int*   y       = (const int*)  d_in[3];   // [2048]
    float* out = (float*)d_out;                     // scalar

    char* w = (char*)d_ws;
    const size_t matBytes = (size_t)NB * KD * sizeof(__bf16);   // 16 MB each
    __bf16* Gbf  = (__bf16*)w;
    __bf16* Xbf  = (__bf16*)(w + matBytes);
    float*  normG = (float*)(w + 2 * matBytes);
    float*  normX = normG + NB;
    int*    pred  = (int*)(normX + NB);

    zero_out_kernel<<<1, 32, 0, stream>>>(out);
    prep_kernel<<<NB, 256, 0, stream>>>(outputs, grad, x, y, Gbf, Xbf,
                                        normG, normX, pred, out);
    pair_kernel<<<NPAIR, 256, LDS_BYTES, stream>>>(Gbf, Xbf, normG, normX, pred, out);
}
